// MambaSimple_29618094473827
// MI455X (gfx1250) — matrix operand order repaired and checked by reading, not compiled, not run on hardware
//
#include <hip/hip_runtime.h>
#include <hip/hip_bf16.h>
#include <math.h>

// ---------------------------------------------------------------------------
// Problem constants (from the reference)
// ---------------------------------------------------------------------------
#define D_MODEL  1024
#define D_STATE  16
#define D_CONV   4
#define D_INNER  2048          // 2 * D_MODEL
#define BATCH    4
#define SEQ      4096
#define NTOK     (BATCH * SEQ)         // 16384 tokens
#define N1       (D_INNER + D_STATE)   // 2064: xc columns + the only 16 z cols that matter
#define N2       (2 * D_STATE + D_STATE)  // 48: [x_dbl(32) | dt_raw(16)]

// Scan chunking
#define CHUNK    128
#define NCHUNK   (SEQ / CHUNK)         // 32
#define NJOB     (BATCH * NCHUNK)      // 128

typedef __attribute__((ext_vector_type(16))) __bf16          v16bf;
typedef __attribute__((ext_vector_type(8)))  float           v8f;
typedef __attribute__((ext_vector_type(8)))  unsigned short  ushort8;

union FragBF { v16bf v; ushort8 h[2]; };

__device__ __forceinline__ unsigned short f2bf(float f) {
    unsigned int u = __builtin_bit_cast(unsigned int, f);
    unsigned int r = u + 0x7FFFu + ((u >> 16) & 1u);   // round-to-nearest-even
    return (unsigned short)(r >> 16);
}

// LDS offset of a generic pointer to __shared__: flat LDS addresses are
// {SHARED_BASE[63:32], lds_offset}, so the low 32 bits are the DS address.
__device__ __forceinline__ unsigned int lds_off(const void* p) {
    return (unsigned int)(unsigned long long)p;
}

// CDNA5 async global->LDS copy (16B per lane), tracked by ASYNCcnt.
__device__ __forceinline__ void async_copy_b128(unsigned int dst_lds,
                                                const void* src_global) {
    asm volatile("global_load_async_to_lds_b128 %0, %1, off"
                 :: "v"(dst_lds), "v"(src_global) : "memory");
}
__device__ __forceinline__ void wait_asynccnt0() {
    asm volatile("s_wait_asynccnt 0x0" ::: "memory");
}

// ---------------------------------------------------------------------------
// f32 -> bf16 conversion (x and w_in rows 0..2063)
// ---------------------------------------------------------------------------
__global__ void cvt_bf16_kernel(const float* __restrict__ src,
                                unsigned short* __restrict__ dst, int n) {
    int i = blockIdx.x * blockDim.x + threadIdx.x;
    if (i < n) dst[i] = f2bf(src[i]);
}

// Pack [w_xproj(32x2048); w_dt(16x2048)] -> bf16 [48x2048], and bias2[48]
__global__ void pack_wp_kernel(const float* __restrict__ w_xproj,
                               const float* __restrict__ w_dt,
                               const float* __restrict__ b_dt,
                               unsigned short* __restrict__ wp,
                               float* __restrict__ bias2) {
    int i = blockIdx.x * blockDim.x + threadIdx.x;
    int nx = 2 * D_STATE * D_INNER;               // 65536
    int nt = nx + D_STATE * D_INNER;              // 98304
    if (i < nt) wp[i] = f2bf(i < nx ? w_xproj[i] : w_dt[i - nx]);
    if (i < N2) bias2[i] = (i < 2 * D_STATE) ? 0.f : b_dt[i - 2 * D_STATE];
}

// ---------------------------------------------------------------------------
// WMMA bf16 GEMM:  C[M,N] = A[M,K] * B[N,K]^T + bias[N]
//   block: 256 threads (8 waves), macro-tile 128(M) x 64(N), K-step 32
//   per wave: 2x2 tiles of 16x16 f32 accumulators
//   Staging: double-buffered global_load_async_to_lds_b128 (ASYNCcnt)
// ---------------------------------------------------------------------------
#define BM 128
#define BN 64
#define BK 32
#define AST 40   // LDS row stride (elems): 32 + 8 pad, keeps 16B alignment
#define BST 40

__global__ __launch_bounds__(256)
void wmma_gemm_kernel(const unsigned short* __restrict__ A, int lda,
                      const unsigned short* __restrict__ B, int ldb,
                      const float* __restrict__ bias,
                      float* __restrict__ C, int ldc,
                      int M, int N, int K) {
    __shared__ __align__(16) unsigned short As[2][BM * AST];   // 2 x 10 KB
    __shared__ __align__(16) unsigned short Bs[2][BN * BST];   // 2 x  5 KB

    const int tid   = threadIdx.x;
    const int lane  = tid & 31;
    const int wave  = tid >> 5;         // 0..7
    const int wm    = wave & 3;         // M quadrant (32 rows each)
    const int wn    = wave >> 2;        // N half     (32 cols each)
    const int lrow  = lane & 15;
    const int lhalf = lane >> 4;        // 0 / 1
    const int bm    = blockIdx.x * BM;
    const int bn    = blockIdx.y * BN;

    const int kA = lhalf * 8;           // A-frag K base within row (ISA 16-bit A layout)
    const int kB = lhalf * 8;           // B-frag K base within row (ISA 16-bit B layout)

    // Per-thread staging coordinates
    const int ar0 = (tid * 2) >> 2;            // A chunk 0 row
    const int ac0 = ((tid * 2) & 3) * 8;       // A chunk 0 col
    const int ar1 = (tid * 2 + 1) >> 2;        // A chunk 1 row
    const int ac1 = ((tid * 2 + 1) & 3) * 8;   // A chunk 1 col
    const int br  = tid >> 2;                  // B row (n) 0..63
    const int bc  = (tid & 3) * 8;             // B col (k)
    const int gnB = bn + br;

    // Issue one K-stage of async global->LDS fills into buffer `buf`.
    auto issue_stage = [&](int k0, int buf) {
        async_copy_b128(lds_off(&As[buf][ar0 * AST + ac0]),
                        A + (size_t)(bm + ar0) * lda + k0 + ac0);
        async_copy_b128(lds_off(&As[buf][ar1 * AST + ac1]),
                        A + (size_t)(bm + ar1) * lda + k0 + ac1);
        if (gnB < N)   // ragged N: skipped rows leave garbage; those cols never stored
            async_copy_b128(lds_off(&Bs[buf][br * BST + bc]),
                            B + (size_t)gnB * ldb + k0 + bc);
    };

    v8f acc[2][2];
#pragma unroll
    for (int i = 0; i < 2; ++i)
#pragma unroll
        for (int j = 0; j < 2; ++j)
#pragma unroll
            for (int r = 0; r < 8; ++r) acc[i][j][r] = 0.f;

    issue_stage(0, 0);
    int cur = 0;

    for (int k0 = 0; k0 < K; k0 += BK) {
        wait_asynccnt0();      // this wave's fills for stage `cur` done
        __syncthreads();       // -> everyone's fills done

        if (k0 + BK < K) issue_stage(k0 + BK, cur ^ 1);   // prefetch next stage

        // --- gather fragments per the gfx1250 16-bit WMMA VGPR layouts ---
        FragBF af[2], bf[2];
#pragma unroll
        for (int i = 0; i < 2; ++i) {
            int row = wm * 32 + i * 16 + lrow;
            af[i].h[0] = *(const ushort8*)(&As[cur][row * AST + kA]);
            af[i].h[1] = *(const ushort8*)(&As[cur][row * AST + kA + 16]);
        }
#pragma unroll
        for (int j = 0; j < 2; ++j) {
            int row = wn * 32 + j * 16 + lrow;
            bf[j].h[0] = *(const ushort8*)(&Bs[cur][row * BST + kB]);
            bf[j].h[1] = *(const ushort8*)(&Bs[cur][row * BST + kB + 16]);
        }

#pragma unroll
        for (int i = 0; i < 2; ++i)
#pragma unroll
            for (int j = 0; j < 2; ++j)
                acc[i][j] = __builtin_amdgcn_wmma_f32_16x16x32_bf16(
                    false, af[i].v, false, bf[j].v,
                    (short)0, acc[i][j], false, false);

        __syncthreads();       // all waves done reading `cur` before its refill
        cur ^= 1;
    }

    // --- epilogue: f32 C/D layout -> global (+bias), guarded on N ---
#pragma unroll
    for (int i = 0; i < 2; ++i)
#pragma unroll
        for (int j = 0; j < 2; ++j) {
            int gn = bn + wn * 32 + j * 16 + lrow;
            if (gn < N) {
                float bsv = bias[gn];
#pragma unroll
                for (int r = 0; r < 8; ++r) {
                    int gm = bm + wm * 32 + i * 16 + r + lhalf * 8;
                    C[(size_t)gm * ldc + gn] = acc[i][j][r] + bsv;
                }
            }
        }
}

// ---------------------------------------------------------------------------
// Depthwise causal conv(4) + SiLU over C1[:, :2048]; emits bf16 xc and f32 xc16
// ---------------------------------------------------------------------------
__global__ void conv_silu_kernel(const float* __restrict__ C1,
                                 const float* __restrict__ w_conv,
                                 const float* __restrict__ b_conv,
                                 unsigned short* __restrict__ xc_bf,
                                 float* __restrict__ xc16) {
    int idx = blockIdx.x * blockDim.x + threadIdx.x;   // t*2048 + c
    if (idx >= NTOK * D_INNER) return;
    int c = idx & (D_INNER - 1);
    int t = idx >> 11;
    int s = t & (SEQ - 1);

    const float* wc = w_conv + c * D_CONV;             // k = 0..3 -> input s-3+k
    const float* base = C1 + (size_t)t * N1 + c;
    float acc = b_conv[c];
    if (s >= 3) acc += wc[0] * base[-3 * N1];
    if (s >= 2) acc += wc[1] * base[-2 * N1];
    if (s >= 1) acc += wc[2] * base[-1 * N1];
    acc += wc[3] * base[0];

    float sv = acc / (1.f + __expf(-acc));             // SiLU
    xc_bf[idx] = f2bf(sv);
    if (c < D_STATE) xc16[t * D_STATE + c] = sv;
}

// ---------------------------------------------------------------------------
// Chunked associative scan for h_t = A_t h_{t-1} + Bx_t   (64 sequences)
// ---------------------------------------------------------------------------
__global__ void scan_chunk_kernel(const float* __restrict__ P,
                                  const float* __restrict__ xc16,
                                  float* __restrict__ Aarr, float* __restrict__ Bxarr,
                                  float* __restrict__ Ac, float* __restrict__ Bc) {
    int job = blockIdx.x;                 // b * NCHUNK + chunk
    int j = threadIdx.x;
    if (j >= D_STATE) return;
    int b = job >> 5, ch = job & (NCHUNK - 1);
    int t0 = b * SEQ + ch * CHUNK;
    float aP = 1.f, bA = 0.f;
    for (int i = 0; i < CHUNK; ++i) {
        int t = t0 + i;
        float xd0 = P[t * N2 + j];                  // x_dbl[:, :16]
        float xd1 = P[t * N2 + D_STATE + j];        // x_dbl[:, 16:32]
        float dtr = P[t * N2 + 2 * D_STATE + j];    // dt_raw (bias already applied)
        float delta = (dtr > 20.f) ? dtr : log1pf(__expf(dtr));   // softplus
        float a  = __expf(-fabsf(xd0) * delta);
        float bx = xd1 * delta * xc16[t * D_STATE + j];
        Aarr[t * D_STATE + j]  = a;
        Bxarr[t * D_STATE + j] = bx;
        aP *= a;
        bA = a * bA + bx;
    }
    Ac[job * D_STATE + j] = aP;
    Bc[job * D_STATE + j] = bA;
}

__global__ void scan_carry_kernel(const float* __restrict__ Ac,
                                  const float* __restrict__ Bc,
                                  float* __restrict__ Hin) {
    int b = blockIdx.x;
    int j = threadIdx.x;
    if (j >= D_STATE) return;
    float h = 0.f;
    for (int ch = 0; ch < NCHUNK; ++ch) {
        int job = b * NCHUNK + ch;
        Hin[job * D_STATE + j] = h;                // carry-in (exclusive)
        h = Ac[job * D_STATE + j] * h + Bc[job * D_STATE + j];
    }
}

__global__ void scan_apply_kernel(const float* __restrict__ Aarr,
                                  const float* __restrict__ Bxarr,
                                  const float* __restrict__ Hin,
                                  const float* __restrict__ C1,
                                  float* __restrict__ y16) {
    int job = blockIdx.x;
    int j = threadIdx.x;
    if (j >= D_STATE) return;
    int b = job >> 5, ch = job & (NCHUNK - 1);
    int t0 = b * SEQ + ch * CHUNK;
    float h = Hin[job * D_STATE + j];
    for (int i = 0; i < CHUNK; ++i) {
        int t = t0 + i;
        h = Aarr[t * D_STATE + j] * h + Bxarr[t * D_STATE + j];
        float z = C1[(size_t)t * N1 + D_INNER + j];     // z[..., :16]
        y16[t * D_STATE + j] = h * z;
    }
}

// ---------------------------------------------------------------------------
// out[m,n] = sum_{k<16} y16[m,k] * w_out[n,k]   (memory-bound, 64MB store)
// ---------------------------------------------------------------------------
__global__ void out_proj_kernel(const float* __restrict__ y16,
                                const float* __restrict__ w_out,
                                float* __restrict__ out) {
    int idx = blockIdx.x * blockDim.x + threadIdx.x;
    if (idx >= NTOK * D_MODEL) return;
    int n = idx & (D_MODEL - 1);
    int m = idx >> 10;
    const float4* yp = (const float4*)(y16 + m * D_STATE);
    const float4* wp = (const float4*)(w_out + (size_t)n * D_INNER);
    float acc = 0.f;
#pragma unroll
    for (int q = 0; q < 4; ++q) {
        float4 a = yp[q], b = wp[q];
        acc += a.x * b.x + a.y * b.y + a.z * b.z + a.w * b.w;
    }
    out[idx] = acc;
}

// ---------------------------------------------------------------------------
// Launch
// ---------------------------------------------------------------------------
extern "C" void kernel_launch(void* const* d_in, const int* in_sizes, int n_in,
                              void* d_out, int out_size, void* d_ws, size_t ws_size,
                              hipStream_t stream) {
    const float* x       = (const float*)d_in[0];
    const float* w_in    = (const float*)d_in[1];
    const float* b_in    = (const float*)d_in[2];
    const float* w_conv  = (const float*)d_in[3];
    const float* b_conv  = (const float*)d_in[4];
    const float* w_xproj = (const float*)d_in[5];
    const float* w_dt    = (const float*)d_in[6];
    const float* b_dt    = (const float*)d_in[7];
    const float* w_out   = (const float*)d_in[8];
    float* out = (float*)d_out;

    // Workspace sub-allocation (256B aligned)
    char* ws = (char*)d_ws;
    size_t off = 0;
    auto take = [&](size_t bytes) -> char* {
        char* p = ws + off;
        off = (off + bytes + 255) & ~(size_t)255;
        return p;
    };
    unsigned short* xbf   = (unsigned short*)take((size_t)NTOK * D_MODEL * 2);   // 32 MB
    unsigned short* w1bf  = (unsigned short*)take((size_t)N1 * D_MODEL * 2);     //  4 MB
    unsigned short* wpbf  = (unsigned short*)take((size_t)N2 * D_INNER * 2);     // 192 KB
    float*          bias2 = (float*)take(N2 * 4);
    float*          C1    = (float*)take((size_t)NTOK * N1 * 4);                 // 135 MB
    unsigned short* xcbf  = (unsigned short*)take((size_t)NTOK * D_INNER * 2);   // 64 MB
    float*          xc16  = (float*)take((size_t)NTOK * D_STATE * 4);            //  1 MB
    float*          P     = (float*)take((size_t)NTOK * N2 * 4);                 //  3 MB
    float*          Aarr  = (float*)take((size_t)NTOK * D_STATE * 4);
    float*          Bxarr = (float*)take((size_t)NTOK * D_STATE * 4);
    float*          Ac    = (float*)take(NJOB * D_STATE * 4);
    float*          Bc    = (float*)take(NJOB * D_STATE * 4);
    float*          Hin   = (float*)take(NJOB * D_STATE * 4);
    float*          y16   = (float*)take((size_t)NTOK * D_STATE * 4);
    (void)ws_size; (void)in_sizes; (void)n_in; (void)out_size;

    // 1. bf16 conversions / packing
    {
        int n = NTOK * D_MODEL;
        cvt_bf16_kernel<<<(n + 255) / 256, 256, 0, stream>>>(x, xbf, n);
    }
    {
        int n = N1 * D_MODEL;                       // first 2064 rows of w_in (contiguous)
        cvt_bf16_kernel<<<(n + 255) / 256, 256, 0, stream>>>(w_in, w1bf, n);
    }
    {
        int n = N2 * D_INNER;
        pack_wp_kernel<<<(n + 255) / 256, 256, 0, stream>>>(w_xproj, w_dt, b_dt, wpbf, bias2);
    }

    // 2. GEMM1: C1[16384, 2064] = x @ w_in[:2064].T + b_in   (xc_lin | z16)
    {
        dim3 grid(NTOK / BM, (N1 + BN - 1) / BN);   // 128 x 33
        wmma_gemm_kernel<<<grid, 256, 0, stream>>>(xbf, D_MODEL, w1bf, D_MODEL,
                                                   b_in, C1, N1, NTOK, N1, D_MODEL);
    }

    // 3. causal depthwise conv + SiLU
    {
        int n = NTOK * D_INNER;
        conv_silu_kernel<<<(n + 255) / 256, 256, 0, stream>>>(C1, w_conv, b_conv, xcbf, xc16);
    }

    // 4. GEMM2: P[16384, 48] = xc @ [w_xproj; w_dt].T + [0; b_dt]
    {
        dim3 grid(NTOK / BM, (N2 + BN - 1) / BN);   // 128 x 1
        wmma_gemm_kernel<<<grid, 256, 0, stream>>>(xcbf, D_INNER, wpbf, D_INNER,
                                                   bias2, P, N2, NTOK, N2, D_INNER);
    }

    // 5. chunked scan
    scan_chunk_kernel<<<NJOB, 32, 0, stream>>>(P, xc16, Aarr, Bxarr, Ac, Bc);
    scan_carry_kernel<<<BATCH, 32, 0, stream>>>(Ac, Bc, Hin);
    scan_apply_kernel<<<NJOB, 32, 0, stream>>>(Aarr, Bxarr, Hin, C1, y16);

    // 6. output projection (only first 16 columns of w_out matter)
    {
        int n = NTOK * D_MODEL;
        out_proj_kernel<<<(n + 255) / 256, 256, 0, stream>>>(y16, w_out, out);
    }
}
